// Attn_30734785970994
// MI455X (gfx1250) — matrix operand order repaired and checked by reading, not compiled, not run on hardware
//
#include <hip/hip_runtime.h>

#define DEV __device__ __forceinline__

typedef __attribute__((ext_vector_type(16))) __bf16 bf16x16;
typedef __attribute__((ext_vector_type(8)))  __bf16 bf16x8;
typedef __attribute__((ext_vector_type(4)))  __bf16 bf16x4;
typedef __attribute__((ext_vector_type(8)))  float  f32x8;

constexpr int  Bc   = 2;
constexpr int  Nn   = 2048;
constexpr int  NCc  = 128;
constexpr int  Dd   = 2048;
constexpr int  Hh   = 16;
constexpr int  DHh  = Dd / Hh;        // 128
constexpr int  Ss   = Nn + NCc;       // 2176
constexpr float SQRT_D   = 45.254833995939045f;   // sqrt(2048)
constexpr float SM_SCALE = 0.08838834764831845f;  // 1/sqrt(128)

// ---------------------------------------------------------------------------
// WMMA fragment loaders per CDNA5 ISA layouts (wave32), vectorized as two
// 16-byte chunks each (b128 loads from LDS or global).
// A (16x32 bf16): lane L holds row (L&15); lanes 0-15 pack K=0..7 / K=16..23,
// lanes 16-31 pack K=8..15 / K=24..31.
DEV bf16x16 load_a(const __bf16* p, int ld, int lane) {
  const __bf16* row = p + (size_t)(lane & 15) * ld + ((lane & 16) ? 8 : 0);
  bf16x8 lo = *(const bf16x8*)(row);       // K = klo .. klo+7
  bf16x8 hi = *(const bf16x8*)(row + 16);  // K = klo+16 .. klo+23
  return __builtin_shufflevector(lo, hi, 0, 1, 2, 3, 4, 5, 6, 7,
                                 8, 9, 10, 11, 12, 13, 14, 15);
}

// B (32x16 bf16), source stored N-major: element B[k][n] = p[n*ld + k].
// lanes 0-15 hold K=0..7 / K=16..23, lanes 16-31 hold K=8..15 / K=24..31.
DEV bf16x16 load_b(const __bf16* p, int ld, int lane) {
  const __bf16* col = p + (size_t)(lane & 15) * ld + ((lane & 16) ? 8 : 0);
  bf16x8 lo = *(const bf16x8*)(col);
  bf16x8 hi = *(const bf16x8*)(col + 16);
  return __builtin_shufflevector(lo, hi, 0, 1, 2, 3, 4, 5, 6, 7,
                                 8, 9, 10, 11, 12, 13, 14, 15);
}

DEV f32x8 wmma_bf16(bf16x16 a, bf16x16 b, f32x8 c) {
  return __builtin_amdgcn_wmma_f32_16x16x32_bf16(
      /*neg_a=*/false, a, /*neg_b=*/false, b,
      /*c_mod=*/(short)0, c, /*reuse_a=*/false, /*reuse_b=*/false);
}

// ---------------------------------------------------------------------------
// Kernel 1: Y[M,Nout](bf16) = X[M,K](f32) @ W[Nout,K](f32)^T
// 128x64 workgroup tile, 4 waves, each wave a 32x64 strip:
// 2 A-frags x 4 B-frags = 8 WMMAs per 32-K step.
__global__ __launch_bounds__(128) void gemm_f32_bf16(
    const float* __restrict__ X, const float* __restrict__ W,
    __bf16* __restrict__ Y, int M, int Nout, int Kd) {
  __shared__ __bf16 As[128][40];
  __shared__ __bf16 Bs[64][40];
  const int n0 = blockIdx.x * 64, m0 = blockIdx.y * 128;
  const int wave = threadIdx.x >> 5, lane = threadIdx.x & 31;

  f32x8 acc[8];
#pragma unroll
  for (int i = 0; i < 8; ++i) acc[i] = (f32x8){};

  const int fr = threadIdx.x >> 3;        // 0..15
  const int fc = (threadIdx.x & 7) * 4;   // 0..28

  for (int kt = 0; kt < Kd; kt += 32) {
#pragma unroll
    for (int i = 0; i < 8; ++i) {         // A tile: 128 x 32
      int r = fr + i * 16;
      float4 xv = *(const float4*)(X + (size_t)(m0 + r) * Kd + kt + fc);
      bf16x4 xb = {(__bf16)xv.x, (__bf16)xv.y, (__bf16)xv.z, (__bf16)xv.w};
      *(bf16x4*)&As[r][fc] = xb;
    }
#pragma unroll
    for (int i = 0; i < 4; ++i) {         // B tile: 64 x 32
      int r = fr + i * 16;
      float4 wv = *(const float4*)(W + (size_t)(n0 + r) * Kd + kt + fc);
      bf16x4 wb = {(__bf16)wv.x, (__bf16)wv.y, (__bf16)wv.z, (__bf16)wv.w};
      *(bf16x4*)&Bs[r][fc] = wb;
    }
    __syncthreads();
    // preload all fragments, then run the 8 WMMAs back-to-back
    bf16x16 a0 = load_a(&As[wave * 32][0], 40, lane);
    bf16x16 a1 = load_a(&As[wave * 32 + 16][0], 40, lane);
    bf16x16 b0 = load_b(&Bs[0][0], 40, lane);
    bf16x16 b1 = load_b(&Bs[16][0], 40, lane);
    bf16x16 b2 = load_b(&Bs[32][0], 40, lane);
    bf16x16 b3 = load_b(&Bs[48][0], 40, lane);
    acc[0] = wmma_bf16(a0, b0, acc[0]);
    acc[1] = wmma_bf16(a0, b1, acc[1]);
    acc[2] = wmma_bf16(a0, b2, acc[2]);
    acc[3] = wmma_bf16(a0, b3, acc[3]);
    acc[4] = wmma_bf16(a1, b0, acc[4]);
    acc[5] = wmma_bf16(a1, b1, acc[5]);
    acc[6] = wmma_bf16(a1, b2, acc[6]);
    acc[7] = wmma_bf16(a1, b3, acc[7]);
    __syncthreads();
  }
  const int n = lane & 15, mb = (lane & 16) ? 8 : 0;
#pragma unroll
  for (int half = 0; half < 2; ++half)
#pragma unroll
    for (int nt = 0; nt < 4; ++nt)
#pragma unroll
      for (int r = 0; r < 8; ++r)
        Y[(size_t)(m0 + wave * 32 + half * 16 + mb + r) * Nout +
          n0 + nt * 16 + n] = (__bf16)acc[half * 4 + nt][r];
}

// ---------------------------------------------------------------------------
// Kernel 2: per (b, s, h): L2-norm q/k over DH, learned scale * sqrt(D), RoPE.
__global__ __launch_bounds__(128) void normrope(
    const __bf16* __restrict__ qkv_x, const __bf16* __restrict__ qkv_c,
    const float* __restrict__ scale, const float* __restrict__ cross_scale,
    __bf16* __restrict__ Q, __bf16* __restrict__ K, __bf16* __restrict__ V) {
  int gid  = blockIdx.x * 4 + (threadIdx.x >> 5);
  int lane = threadIdx.x & 31;
  int h = gid % Hh;
  int s = (gid / Hh) % Ss;
  int b = gid / (Hh * Ss);
  if (b >= Bc) return;

  const __bf16* row;
  const float*  sc;
  if (s < Nn) { row = qkv_x + (size_t)(b * Nn + s) * (3 * Dd); sc = scale; }
  else        { row = qkv_c + (size_t)(b * NCc + (s - Nn)) * (3 * Dd); sc = cross_scale; }

  float q[4], k[4], v[4];
#pragma unroll
  for (int j = 0; j < 4; ++j) {
    int d = lane + j * 32;
    q[j] = (float)row[h * DHh + d];
    k[j] = (float)row[Dd + h * DHh + d];
    v[j] = (float)row[2 * Dd + h * DHh + d];
  }
  float qs = q[0] * q[0] + q[1] * q[1] + q[2] * q[2] + q[3] * q[3];
  float ks = k[0] * k[0] + k[1] * k[1] + k[2] * k[2] + k[3] * k[3];
#pragma unroll
  for (int m = 16; m >= 1; m >>= 1) {
    qs += __shfl_xor(qs, m);
    ks += __shfl_xor(ks, m);
  }
  float qinv = 1.0f / fmaxf(sqrtf(qs), 1e-12f);
  float kinv = 1.0f / fmaxf(sqrtf(ks), 1e-12f);
#pragma unroll
  for (int j = 0; j < 4; ++j) {
    int d = lane + j * 32;
    float g = sc[h * DHh + d] * SQRT_D;
    q[j] *= qinv * g;
    k[j] *= kinv * g;
  }
  // RoPE: pair (i, i+64); lane's pairs are (q[0],q[2]) at i=lane, (q[1],q[3]) at i=lane+32.
  float f0 = (float)s * __powf(10000.0f, -(float)lane * (1.0f / 64.0f));
  float f1 = (float)s * __powf(10000.0f, -(float)(lane + 32) * (1.0f / 64.0f));
  float c0 = cosf(f0), s0 = sinf(f0), c1 = cosf(f1), s1 = sinf(f1);
  float qo0 = q[0] * c0 - q[2] * s0, qo2 = q[0] * s0 + q[2] * c0;
  float qo1 = q[1] * c1 - q[3] * s1, qo3 = q[1] * s1 + q[3] * c1;
  float ko0 = k[0] * c0 - k[2] * s0, ko2 = k[0] * s0 + k[2] * c0;
  float ko1 = k[1] * c1 - k[3] * s1, ko3 = k[1] * s1 + k[3] * c1;

  size_t o = (((size_t)b * Hh + h) * Ss + s) * (size_t)DHh;
  Q[o + lane]      = (__bf16)qo0;  Q[o + lane + 32] = (__bf16)qo1;
  Q[o + lane + 64] = (__bf16)qo2;  Q[o + lane + 96] = (__bf16)qo3;
  K[o + lane]      = (__bf16)ko0;  K[o + lane + 32] = (__bf16)ko1;
  K[o + lane + 64] = (__bf16)ko2;  K[o + lane + 96] = (__bf16)ko3;
#pragma unroll
  for (int j = 0; j < 4; ++j) V[o + lane + j * 32] = (__bf16)v[j];
}

// ---------------------------------------------------------------------------
// Kernel 3: flash attention. grid(B*H, N/64), 4 waves, 16 query rows per wave,
// kv in blocks of 32. K and transposed V staged cooperatively in LDS per block.
__global__ __launch_bounds__(128) void attn(
    const __bf16* __restrict__ Q, const __bf16* __restrict__ K,
    const __bf16* __restrict__ V, __bf16* __restrict__ O) {
  const int bh = blockIdx.x;
  const int b = bh / Hh, h = bh % Hh;
  const int wave = threadIdx.x >> 5, lane = threadIdx.x & 31;
  const int q0 = blockIdx.y * 64 + wave * 16;

  const __bf16* Qp = Q + ((size_t)bh * Ss + q0) * DHh;
  const __bf16* Kp = K + (size_t)bh * Ss * DHh;
  const __bf16* Vp = V + (size_t)bh * Ss * DHh;

  __shared__ __bf16 Ks[32][136];      // K tile, row = kv, col = d
  __shared__ __bf16 Vt[128][40];      // V tile transposed: Vt[d][kv]
  __shared__ __bf16 Plds[4][16][40];  // per-wave P tile (16 q x 32 kv)

  f32x8 acc[8];
#pragma unroll
  for (int i = 0; i < 8; ++i) acc[i] = (f32x8){};
  float rowmax[8], rowsum[8];
#pragma unroll
  for (int r = 0; r < 8; ++r) { rowmax[r] = -1e30f; rowsum[r] = 0.0f; }

  bf16x16 qfrag[4];
#pragma unroll
  for (int kc = 0; kc < 4; ++kc) qfrag[kc] = load_a(Qp + kc * 32, DHh, lane);

  const int n = lane & 15, mb = (lane & 16) ? 8 : 0;
  const int tr = threadIdx.x >> 4;        // 0..7
  const int tc = (threadIdx.x & 15) * 8;  // 0..120

  for (int kv = 0; kv < Ss; kv += 32) {
    // cooperative stage: K (direct) and V (transposed) for this kv block
#pragma unroll
    for (int i = 0; i < 4; ++i) {
      int r = tr + i * 8;  // kv row in block
      *(bf16x8*)&Ks[r][tc] =
          *(const bf16x8*)(Kp + (size_t)(kv + r) * DHh + tc);
      bf16x8 vv = *(const bf16x8*)(Vp + (size_t)(kv + r) * DHh + tc);
#pragma unroll
      for (int j = 0; j < 8; ++j) Vt[tc + j][r] = vv[j];
    }
    __syncthreads();

    f32x8 sc0 = (f32x8){}, sc1 = (f32x8){};
#pragma unroll
    for (int kc = 0; kc < 4; ++kc) {
      bf16x16 b0 = load_b(&Ks[0][kc * 32], 136, lane);
      bf16x16 b1 = load_b(&Ks[16][kc * 32], 136, lane);
      sc0 = wmma_bf16(qfrag[kc], b0, sc0);
      sc1 = wmma_bf16(qfrag[kc], b1, sc1);
    }
#pragma unroll
    for (int r = 0; r < 8; ++r) {
      float e0 = sc0[r] * SM_SCALE, e1 = sc1[r] * SM_SCALE;
      float m = fmaxf(e0, e1);
      m = fmaxf(m, __shfl_xor(m, 1));
      m = fmaxf(m, __shfl_xor(m, 2));
      m = fmaxf(m, __shfl_xor(m, 4));
      m = fmaxf(m, __shfl_xor(m, 8));
      float mnew  = fmaxf(rowmax[r], m);
      float alpha = __expf(rowmax[r] - mnew);
      rowmax[r] = mnew;
      float p0 = __expf(e0 - mnew), p1 = __expf(e1 - mnew);
      float ps = p0 + p1;
      ps += __shfl_xor(ps, 1);
      ps += __shfl_xor(ps, 2);
      ps += __shfl_xor(ps, 4);
      ps += __shfl_xor(ps, 8);
      rowsum[r] = rowsum[r] * alpha + ps;
      Plds[wave][mb + r][n]      = (__bf16)p0;
      Plds[wave][mb + r][n + 16] = (__bf16)p1;
#pragma unroll
      for (int dt = 0; dt < 8; ++dt) acc[dt][r] *= alpha;
    }
    bf16x16 pa = load_a(&Plds[wave][0][0], 40, lane);
#pragma unroll
    for (int dt = 0; dt < 8; ++dt) {
      bf16x16 vb = load_b(&Vt[dt * 16][0], 40, lane);
      acc[dt] = wmma_bf16(pa, vb, acc[dt]);
    }
    __syncthreads();
  }
#pragma unroll
  for (int dt = 0; dt < 8; ++dt)
#pragma unroll
    for (int r = 0; r < 8; ++r)
      O[((size_t)b * Nn + q0 + mb + r) * Dd + h * DHh + dt * 16 + n] =
          (__bf16)(acc[dt][r] / rowsum[r]);
}

// ---------------------------------------------------------------------------
// Kernel 4: out[M,Nout](f32) = X[M,K](bf16) @ W[Nout,K](f32)^T + bias
// Same 128x64 tiling as kernel 1.
__global__ __launch_bounds__(128) void gemm_out(
    const __bf16* __restrict__ X, const float* __restrict__ W,
    const float* __restrict__ bias, float* __restrict__ Y,
    int M, int Nout, int Kd) {
  __shared__ __bf16 As[128][40];
  __shared__ __bf16 Bs[64][40];
  const int n0 = blockIdx.x * 64, m0 = blockIdx.y * 128;
  const int wave = threadIdx.x >> 5, lane = threadIdx.x & 31;

  f32x8 acc[8];
#pragma unroll
  for (int i = 0; i < 8; ++i) acc[i] = (f32x8){};

  const int ar = threadIdx.x >> 2;        // 0..31
  const int ac = (threadIdx.x & 3) * 8;   // 0..24
  const int fr = threadIdx.x >> 3;        // 0..15
  const int fc = (threadIdx.x & 7) * 4;   // 0..28

  for (int kt = 0; kt < Kd; kt += 32) {
#pragma unroll
    for (int i = 0; i < 4; ++i) {         // A tile: 128 x 32 (bf16 source)
      int r = ar + i * 32;
      *(bf16x8*)&As[r][ac] =
          *(const bf16x8*)(X + (size_t)(m0 + r) * Kd + kt + ac);
    }
#pragma unroll
    for (int i = 0; i < 4; ++i) {         // B tile: 64 x 32 (f32 source)
      int r = fr + i * 16;
      float4 wv = *(const float4*)(W + (size_t)(n0 + r) * Kd + kt + fc);
      bf16x4 wb = {(__bf16)wv.x, (__bf16)wv.y, (__bf16)wv.z, (__bf16)wv.w};
      *(bf16x4*)&Bs[r][fc] = wb;
    }
    __syncthreads();
    bf16x16 a0 = load_a(&As[wave * 32][0], 40, lane);
    bf16x16 a1 = load_a(&As[wave * 32 + 16][0], 40, lane);
    bf16x16 b0 = load_b(&Bs[0][0], 40, lane);
    bf16x16 b1 = load_b(&Bs[16][0], 40, lane);
    bf16x16 b2 = load_b(&Bs[32][0], 40, lane);
    bf16x16 b3 = load_b(&Bs[48][0], 40, lane);
    acc[0] = wmma_bf16(a0, b0, acc[0]);
    acc[1] = wmma_bf16(a0, b1, acc[1]);
    acc[2] = wmma_bf16(a0, b2, acc[2]);
    acc[3] = wmma_bf16(a0, b3, acc[3]);
    acc[4] = wmma_bf16(a1, b0, acc[4]);
    acc[5] = wmma_bf16(a1, b1, acc[5]);
    acc[6] = wmma_bf16(a1, b2, acc[6]);
    acc[7] = wmma_bf16(a1, b3, acc[7]);
    __syncthreads();
  }
  const int n = lane & 15, mb = (lane & 16) ? 8 : 0;
#pragma unroll
  for (int half = 0; half < 2; ++half)
#pragma unroll
    for (int nt = 0; nt < 4; ++nt)
#pragma unroll
      for (int r = 0; r < 8; ++r)
        Y[(size_t)(m0 + wave * 32 + half * 16 + mb + r) * Nout +
          n0 + nt * 16 + n] =
            acc[half * 4 + nt][r] + bias[n0 + nt * 16 + n];
}

// ---------------------------------------------------------------------------
extern "C" void kernel_launch(void* const* d_in, const int* in_sizes, int n_in,
                              void* d_out, int out_size, void* d_ws, size_t ws_size,
                              hipStream_t stream) {
  (void)in_sizes; (void)n_in; (void)out_size; (void)ws_size;
  const float* x        = (const float*)d_in[0];
  const float* c        = (const float*)d_in[1];
  const float* w_qkv    = (const float*)d_in[2];
  const float* w_cqkv   = (const float*)d_in[3];
  const float* w_out    = (const float*)d_in[4];
  const float* b_out    = (const float*)d_in[5];
  const float* scale    = (const float*)d_in[6];
  const float* cscale   = (const float*)d_in[7];
  float* out = (float*)d_out;

  // workspace carve-up (bf16 staging buffers)
  char* ws = (char*)d_ws;
  size_t off = 0;
  __bf16* qkv_x = (__bf16*)(ws + off); off += (size_t)Bc * Nn * 3 * Dd * sizeof(__bf16);
  __bf16* qkv_c = (__bf16*)(ws + off); off += (size_t)Bc * NCc * 3 * Dd * sizeof(__bf16);
  __bf16* Qb    = (__bf16*)(ws + off); off += (size_t)Bc * Hh * Ss * DHh * sizeof(__bf16);
  __bf16* Kb    = (__bf16*)(ws + off); off += (size_t)Bc * Hh * Ss * DHh * sizeof(__bf16);
  __bf16* Vb    = (__bf16*)(ws + off); off += (size_t)Bc * Hh * Ss * DHh * sizeof(__bf16);
  __bf16* Ob    = (__bf16*)(ws + off); off += (size_t)Bc * Nn * Dd * sizeof(__bf16);

  dim3 blk(128);
  // 1) QKV projections (self + cross)
  gemm_f32_bf16<<<dim3((3 * Dd) / 64, (Bc * Nn) / 128), blk, 0, stream>>>(
      x, w_qkv, qkv_x, Bc * Nn, 3 * Dd, Dd);
  gemm_f32_bf16<<<dim3((3 * Dd) / 64, (Bc * NCc) / 128), blk, 0, stream>>>(
      c, w_cqkv, qkv_c, Bc * NCc, 3 * Dd, Dd);
  // 2) QK-norm + scale + RoPE, scatter into (B,H,S,DH)
  normrope<<<dim3((Bc * Ss * Hh) / 4), blk, 0, stream>>>(
      qkv_x, qkv_c, scale, cscale, Qb, Kb, Vb);
  // 3) flash attention over S=2176, self queries only
  attn<<<dim3(Bc * Hh, Nn / 64), blk, 0, stream>>>(Qb, Kb, Vb, Ob);
  // 4) output projection + bias
  gemm_out<<<dim3(Dd / 64, (Bc * Nn) / 128), blk, 0, stream>>>(
      Ob, w_out, b_out, out, Bc * Nn, Dd, Dd);
}
